// coGN_78709570666652
// MI455X (gfx1250) — matrix operand order repaired and checked by reading, not compiled, not run on hardware
//
#include <hip/hip_runtime.h>
#include <stdint.h>

// ============================================================================
// coGN-style GNN forward for MI455X (gfx1250).
// All GEMMs via v_wmma_f32_16x16x32_f16 (f16 operands, f32 accum).
// Weight tiles staged into LDS by the Tensor Data Mover (tensor_load_to_lds,
// TENSORcnt) with hardware row padding; indexed activation gathers staged via
// GLOBAL_LOAD_ASYNC_TO_LDS_B128 (ASYNCcnt). fp32 master state for residuals.
// ============================================================================

#define EMB     128
#define NLAYER  5
#define NHID    4
#define NGRAPH  256
#define EPB     64      // edges per block in the MLP kernel (4 M-tiles / wave)

typedef __attribute__((ext_vector_type(16))) _Float16 v16h;
typedef __attribute__((ext_vector_type(8)))  _Float16 v8h;
typedef __attribute__((ext_vector_type(8)))  float    v8f;
typedef __attribute__((ext_vector_type(4)))  int      v4i;
typedef __attribute__((ext_vector_type(8)))  int      v8i;
typedef __attribute__((ext_vector_type(4)))  unsigned int v4u;

typedef __attribute__((address_space(1))) v4i      GV4;
typedef __attribute__((address_space(3))) v4i      LV4;
typedef __attribute__((address_space(3))) _Float16 LDSH;

#if __has_builtin(__builtin_amdgcn_global_load_async_to_lds_b128)
#define HAVE_ASYNC_LDS 1
#else
#define HAVE_ASYNC_LDS 0
#endif

#if __has_builtin(__builtin_amdgcn_tensor_load_to_lds)
#define HAVE_TDM 1
#else
#define HAVE_TDM 0
#endif

// 16-byte global->LDS copy. Async DMA path on gfx1250 (ASYNCcnt tracked),
// synchronous vector copy otherwise.
__device__ __forceinline__ void cp16(_Float16* dst_lds, const _Float16* src_glob) {
#if HAVE_ASYNC_LDS
  __builtin_amdgcn_global_load_async_to_lds_b128((GV4*)src_glob, (LV4*)dst_lds, 0, 0);
#else
  *(v8h*)dst_lds = *(const v8h*)src_glob;
#endif
}
__device__ __forceinline__ void cp_wait() {
#if HAVE_ASYNC_LDS
#if __has_builtin(__builtin_amdgcn_s_wait_asynccnt)
  __builtin_amdgcn_s_wait_asynccnt(0);
#else
  asm volatile("s_wait_asynccnt 0x0" ::: "memory");
#endif
#endif
}

// ---------------------------------------------------------------------------
// TDM weight stager: one tensor_load_to_lds moves a 128-row x 128-half f16
// tile (k-slice of a [128][Kstride] image) into LDS with a 136-half row
// stride produced by descriptor padding (row = 64 DWORDs -> pad_interval=5;
// pad = 4 DWORDs -> pad_amount=3). Issued once per block (TDM ignores EXEC).
// D# packing per CDNA5 ISA 8.3/8.4. This toolchain's builtin is the 6-arg
// form: (u32x4 g0, i32x8 g1, i32x4 g2, i32x4 g3, i32x8 pad, i32 cpol).
// Fallback: per-lane async b128 copies.
// ---------------------------------------------------------------------------
__device__ __forceinline__ void stage_weight_tile(const _Float16* wsrc,
                                                  int Kstride,
                                                  _Float16* sWbase, int t) {
#if HAVE_TDM
  if (t == 0) {
    unsigned long long ga = (unsigned long long)(uintptr_t)wsrc;
    unsigned lds_addr = (unsigned)(uintptr_t)(LDSH*)sWbase;
    v4u g0;
    g0[0] = 1u;                                        // count=1 (valid D#)
    g0[1] = lds_addr;                                  // lds_addr (bytes)
    g0[2] = (unsigned)(ga & 0xffffffffu);              // global_addr[31:0]
    g0[3] = (unsigned)((ga >> 32) & 0x01ffffffu)       // global_addr[56:32]
            | (2u << 30);                              // type=2 (image)
    v8i g1;
    g1[0] = (1 << 16)                                  // data_size=2B
          | (1 << 20)                                  // pad_enable
          | (5 << 22)                                  // pad_interval: 64 DWORDs
          | (3 << 25);                                 // pad_amount: 4 DWORDs
    g1[1] = (int)(((unsigned)Kstride & 0xffffu) << 16);            // tensor_dim0 lo
    g1[2] = (int)((((unsigned)Kstride >> 16) & 0xffffu) | (128u << 16)); // dim0 hi | dim1 lo
    g1[3] = (int)(128u << 16);                         // dim1 hi(0) | tile_dim0=128
    g1[4] = 128;                                       // tile_dim1=128, tile_dim2=0
    g1[5] = Kstride;                                   // tensor_dim0_stride lo32
    g1[6] = 0;
    g1[7] = 0;
    v4i z4 = {0, 0, 0, 0};
    v8i z8 = {0, 0, 0, 0, 0, 0, 0, 0};
    __builtin_amdgcn_tensor_load_to_lds(g0, g1, z4, z4, z8, 0);
  }
#else
  int n = t >> 1, kh = (t & 1) * 64;
  const _Float16* p = wsrc + (long)n * Kstride + kh;
#pragma unroll
  for (int j = 0; j < 8; ++j) cp16(sWbase + n * 136 + kh + j * 8, p + j * 8);
#endif
}

__device__ __forceinline__ void tdm_wait(int t) {
#if HAVE_TDM
  if (t == 0) {
#if __has_builtin(__builtin_amdgcn_s_wait_tensorcnt)
    __builtin_amdgcn_s_wait_tensorcnt(0);
#else
    asm volatile("s_wait_tensorcnt 0x0" ::: "memory");
#endif
  }
#endif
}

__device__ __forceinline__ float silu_f(float x) { return x / (1.0f + __expf(-x)); }

// ---------------------------------------------------------------------------
// WMMA fragment helpers (wave32); A/B use contiguous k = kbase + 16*(lane/16) + j.
// ---------------------------------------------------------------------------
__device__ __forceinline__ v16h load_a_frag(const _Float16* A, int lda,
                                            int mtile, int kbase, int lane) {
  int m  = mtile * 16 + (lane & 15);
  int kb = kbase + ((lane & 16) ? 16 : 0);
  const _Float16* p = A + m * lda + kb;
  v16h a;
#pragma unroll
  for (int j = 0; j < 16; ++j) a[j] = p[j];
  return a;
}

// Wt stored transposed in LDS: Wt[n][k], row stride ldw.
__device__ __forceinline__ v16h load_b_frag(const _Float16* Wt, int ldw,
                                            int ntile, int kbase, int lane) {
  int n  = ntile * 16 + (lane & 15);
  int kb = kbase + ((lane & 16) ? 16 : 0);
  const _Float16* p = Wt + n * ldw + kb;
  v16h b;
#pragma unroll
  for (int j = 0; j < 16; ++j) b[j] = p[j];
  return b;
}

__device__ __forceinline__ v8f wmma_f16(v16h a, v16h b, v8f c) {
  return __builtin_amdgcn_wmma_f32_16x16x32_f16(false, a, false, b,
                                                (short)0, c, false, false);
}

__device__ __forceinline__ void store_act(const v8f* acc, int nmt, const float* bias,
                                          _Float16 (*out)[136], int wave, int lane) {
  int   n  = wave * 16 + (lane & 15);
  int   hi = (lane & 16) ? 8 : 0;
  float bv = bias[n];
  for (int mt = 0; mt < nmt; ++mt) {
#pragma unroll
    for (int r = 0; r < 8; ++r)
      out[mt * 16 + r + hi][n] = (_Float16)silu_f(acc[mt][r] + bv);
  }
}

// ============================================================================
// Weight prep: fp32 row-major [K][128] -> f16 transposed [128][K]
// ============================================================================
__global__ __launch_bounds__(256) void transpose_w_kernel(
    const float* __restrict__ W, _Float16* __restrict__ out, int K) {
  long i = (long)blockIdx.x * 256 + threadIdx.x;
  if (i < (long)K * EMB) {
    int k = (int)(i >> 7);
    int n = (int)(i & 127);
    out[(long)n * K + k] = (_Float16)W[i];
  }
}

// ============================================================================
// Node GEMM: mode 0: h_node = atom_table[z] @ W + b
//            mode 1: h_node += silu(agg @ W + b)
// Writes f32 master + f16 shadow. Weights TDM-staged from f16 image.
// ============================================================================
__global__ __launch_bounds__(256) void node_gemm_kernel(
    const float* __restrict__ Asrc, const int* __restrict__ z,
    const _Float16* __restrict__ WT, const float* __restrict__ bias,
    float* __restrict__ h_node, _Float16* __restrict__ h_node_h,
    int N, int mode) {
  __shared__ _Float16 sA[64][136];
  __shared__ _Float16 sW[128][136];   // transposed: sW[n][k]
  int  t = threadIdx.x, lane = t & 31, wave = t >> 5;
  long n0 = (long)blockIdx.x * 64;

  stage_weight_tile(WT, EMB, &sW[0][0], t);
  {   // A rows: fp32 source (gathered for mode 0) -> f16
    int  r = t >> 2, c0 = (t & 3) * 32;
    long node = n0 + r;
    const float* src = nullptr;
    if (node < N)
      src = (mode == 0) ? (Asrc + (long)z[node] * EMB) : (Asrc + node * EMB);
#pragma unroll
    for (int c = 0; c < 32; ++c)
      sA[r][c0 + c] = (_Float16)(src ? src[c0 + c] : 0.0f);
  }
  cp_wait();
  tdm_wait(t);
  __syncthreads();

  v8f acc[4] = {};
#pragma unroll
  for (int ks = 0; ks < 4; ++ks) {
    v16h b = load_b_frag(&sW[0][0], 136, wave, ks * 32, lane);
#pragma unroll
    for (int mt = 0; mt < 4; ++mt) {
      v16h a = load_a_frag(&sA[0][0], 136, mt, ks * 32, lane);
      acc[mt] = wmma_f16(a, b, acc[mt]);
    }
  }

  int   n  = wave * 16 + (lane & 15);
  int   hi = (lane & 16) ? 8 : 0;
  float bv = bias[n];
#pragma unroll
  for (int mt = 0; mt < 4; ++mt) {
#pragma unroll
    for (int r = 0; r < 8; ++r) {
      long node = n0 + mt * 16 + r + hi;
      if (node < N) {
        float x = acc[mt][r] + bv;
        float v = (mode == 0) ? x : (h_node[node * EMB + n] + silu_f(x));
        h_node[node * EMB + n]   = v;
        h_node_h[node * EMB + n] = (_Float16)v;
      }
    }
  }
}

// ============================================================================
// Edge embedding: gaussian basis (K=32, one WMMA k-step) @ edge_emb_w + b
// ============================================================================
__global__ __launch_bounds__(256) void edge_embed_kernel(
    const float* __restrict__ pos, const int* __restrict__ row,
    const int* __restrict__ col, const _Float16* __restrict__ ewT,
    const float* __restrict__ eb, float* __restrict__ h_edge,
    _Float16* __restrict__ h_edge_h, int E) {
  __shared__ float    sD[64];
  __shared__ _Float16 sB[64][40];
  __shared__ _Float16 sW[128][40];   // transposed: sW[n][k], k<32
  int  t = threadIdx.x, lane = t & 31, wave = t >> 5;
  long e0 = (long)blockIdx.x * 64;

  {   // W: async 16B copies from f16 [128][32]
    int n = t >> 1, kh = (t & 1) * 16;
    const _Float16* wsrc = ewT + (long)n * 32 + kh;
#pragma unroll
    for (int j = 0; j < 2; ++j) cp16(&sW[n][kh + j * 8], wsrc + j * 8);
  }
  if (t < 64) {
    long  e = e0 + t;
    float d = 0.0f;
    if (e < E) {
      int   r = row[e], c = col[e];
      float dx = pos[r * 3 + 0] - pos[c * 3 + 0];
      float dy = pos[r * 3 + 1] - pos[c * 3 + 1];
      float dz = pos[r * 3 + 2] - pos[c * 3 + 2];
      d = sqrtf(dx * dx + dy * dy + dz * dz);
    }
    sD[t] = d;
  }
  cp_wait();
  __syncthreads();
  {   // gaussian basis, linspace(0, CUT, BINS), sigma = CUT/BINS
    int   r = t >> 2, c0 = (t & 3) * 8;
    float d = sD[r];
    const float sigma = 5.0f / 32.0f;
    const float step  = 5.0f / 31.0f;
#pragma unroll
    for (int c = 0; c < 8; ++c) {
      float mu = (float)(c0 + c) * step;
      float u  = (d - mu) / sigma;
      sB[r][c0 + c] = (_Float16)__expf(-0.5f * u * u);
    }
  }
  __syncthreads();

  v8f  acc[4] = {};
  v16h b = load_b_frag(&sW[0][0], 40, wave, 0, lane);
#pragma unroll
  for (int mt = 0; mt < 4; ++mt) {
    v16h a = load_a_frag(&sB[0][0], 40, mt, 0, lane);
    acc[mt] = wmma_f16(a, b, acc[mt]);
  }

  int   n  = wave * 16 + (lane & 15);
  int   hi = (lane & 16) ? 8 : 0;
  float bv = eb[n];
#pragma unroll
  for (int mt = 0; mt < 4; ++mt) {
#pragma unroll
    for (int r = 0; r < 8; ++r) {
      long e = e0 + mt * 16 + r + hi;
      if (e < E) {
        float v = acc[mt][r] + bv;
        h_edge[e * EMB + n]   = v;
        h_edge_h[e * EMB + n] = (_Float16)v;
      }
    }
  }
}

// ============================================================================
// Edge MLP (~180 GFLOP/layer). Stage 0 streams the 384-wide K as 3 chunks
// whose A-tiles are h_edge_h | h_node_h[row] | h_node_h[col] — the concat is
// never materialized. Weights: TDM tile loads. A-tiles: async b128 gathers.
// ============================================================================
__global__ __launch_bounds__(256) void edge_mlp_kernel(
    const _Float16* __restrict__ h_node_h, float* __restrict__ h_edge,
    _Float16* __restrict__ h_edge_h,
    const int* __restrict__ row, const int* __restrict__ col,
    const _Float16* __restrict__ w0T, const float* __restrict__ b0,
    const _Float16* __restrict__ whT, const float* __restrict__ bh,
    float* __restrict__ agg, int E) {
  __shared__ _Float16 sA[EPB][136];
  __shared__ _Float16 sW[128][136];          // transposed: sW[n][k]
  __shared__ _Float16 sAct[2][EPB][136];
  int  t = threadIdx.x, lane = t & 31, wave = t >> 5;
  long e0 = (long)blockIdx.x * EPB;

  __builtin_prefetch(w0T, 0, 0);             // global_prefetch: L2-resident weights
  __builtin_prefetch(whT, 0, 0);

  // ---- stage 0: [EPB x 384] @ [384 x 128] over 3 K-chunks of 128 ----
  v8f acc[4] = {};
  for (int chunk = 0; chunk < 3; ++chunk) {
    stage_weight_tile(w0T + chunk * 128, 384, &sW[0][0], t);
    {   // A chunk: per-edge gather, async 16B copies of f16 rows
      int  r = t >> 2, c0 = (t & 3) * 32;
      long e = e0 + r;
      if (e >= E) e = E - 1;                 // tail rows: harmless clamp
      const _Float16* src;
      if (chunk == 0)      src = h_edge_h + e * EMB;
      else if (chunk == 1) src = h_node_h + (long)row[e] * EMB;
      else                 src = h_node_h + (long)col[e] * EMB;
#pragma unroll
      for (int j = 0; j < 4; ++j) cp16(&sA[r][c0 + j * 8], src + c0 + j * 8);
    }
    cp_wait();
    tdm_wait(t);
    __syncthreads();
#pragma unroll
    for (int ks = 0; ks < 4; ++ks) {
      v16h b = load_b_frag(&sW[0][0], 136, wave, ks * 32, lane);
#pragma unroll
      for (int mt = 0; mt < 4; ++mt) {
        v16h a = load_a_frag(&sA[0][0], 136, mt, ks * 32, lane);
        acc[mt] = wmma_f16(a, b, acc[mt]);
      }
    }
    __syncthreads();
  }
  store_act(acc, 4, b0, sAct[0], wave, lane);

  // ---- hidden stages: 4 x [EPB x 128] @ [128 x 128], ping-pong LDS ----
  for (int kH = 0; kH < NHID; ++kH) {
    stage_weight_tile(whT + (long)kH * EMB * EMB, EMB, &sW[0][0], t);
    cp_wait();
    tdm_wait(t);
    __syncthreads();   // sW ready + previous activation writes visible

    const _Float16* inBuf = &sAct[kH & 1][0][0];
    v8f acc2[4] = {};
#pragma unroll
    for (int ks = 0; ks < 4; ++ks) {
      v16h b = load_b_frag(&sW[0][0], 136, wave, ks * 32, lane);
#pragma unroll
      for (int mt = 0; mt < 4; ++mt) {
        v16h a = load_a_frag(inBuf, 136, mt, ks * 32, lane);
        acc2[mt] = wmma_f16(a, b, acc2[mt]);
      }
    }
    store_act(acc2, 4, bh + kH * EMB, sAct[(kH + 1) & 1], wave, lane);
    __syncthreads();
  }

  // ---- epilogue: residual update (fp32 master + f16 shadow) + scatter ----
  {
    int  r = t >> 2, c0 = (t & 3) * 32;
    long e = e0 + r;
    if (e < E) {
      float*     he  = h_edge   + e * EMB;
      _Float16*  heh = h_edge_h + e * EMB;
      float*     ag  = agg + (long)row[e] * EMB;
#pragma unroll
      for (int c = 0; c < 32; ++c) {
        float v = he[c0 + c] + (float)sAct[NHID & 1][r][c0 + c];
        he[c0 + c]  = v;
        heh[c0 + c] = (_Float16)v;
        atomicAdd(&ag[c0 + c], v);
      }
    }
  }
}

// ============================================================================
// Graph pooling + output head
// ============================================================================
__global__ __launch_bounds__(256) void pool_kernel(
    const float* __restrict__ h_node, const int* __restrict__ batch,
    float* __restrict__ sums, float* __restrict__ cnt, int N) {
  long id = (long)blockIdx.x * 256 + threadIdx.x;
  long n  = id >> 2;
  int  q  = (int)(id & 3);
  if (n < N) {
    int          g   = batch[n];
    const float* src = h_node + n * EMB;
    float*       dst = sums + (long)g * EMB;
    int c0 = q * 32;
    for (int c = 0; c < 32; ++c) atomicAdd(&dst[c0 + c], src[c0 + c]);
    if (q == 0) atomicAdd(&cnt[g], 1.0f);
  }
}

__global__ __launch_bounds__(256) void out_kernel(
    const float* __restrict__ sums, const float* __restrict__ cnt,
    const float* __restrict__ out_w, const float* __restrict__ out_b,
    float* __restrict__ out) {
  int   g = threadIdx.x;
  float c = cnt[g];
  if (c < 1.0f) c = 1.0f;
  float acc = 0.0f;
  for (int k = 0; k < EMB; ++k) acc += (sums[g * EMB + k] / c) * out_w[k];
  out[g] = acc + out_b[0];
}

__global__ void zero_kernel(float* __restrict__ p, long n) {
  long i = (long)blockIdx.x * 256 + threadIdx.x;
  if (i < n) p[i] = 0.0f;
}

// ============================================================================
extern "C" void kernel_launch(void* const* d_in, const int* in_sizes, int n_in,
                              void* d_out, int out_size, void* d_ws, size_t ws_size,
                              hipStream_t stream) {
  const int*   z          = (const int*)d_in[0];
  const float* pos        = (const float*)d_in[1];
  const int*   batch      = (const int*)d_in[2];
  const int*   ei         = (const int*)d_in[3];
  const float* atom_table = (const float*)d_in[4];
  const float* atom_w     = (const float*)d_in[5];
  const float* atom_b     = (const float*)d_in[6];
  const float* edge_emb_w = (const float*)d_in[7];
  const float* edge_emb_b = (const float*)d_in[8];
  const float* edge_w0    = (const float*)d_in[9];
  const float* edge_b0    = (const float*)d_in[10];
  const float* edge_w     = (const float*)d_in[11];
  const float* edge_b     = (const float*)d_in[12];
  const float* node_w     = (const float*)d_in[13];
  const float* node_b     = (const float*)d_in[14];
  const float* out_w      = (const float*)d_in[15];
  const float* out_b      = (const float*)d_in[16];

  int N = in_sizes[0];
  int E = in_sizes[3] / 2;
  const int* row = ei;
  const int* col = ei + E;

  // ---- workspace carve-up ----
  char*  ws     = (char*)d_ws;
  float* h_node = (float*)ws;  ws += (size_t)N * EMB * 4;
  float* agg    = (float*)ws;  ws += (size_t)N * EMB * 4;
  float* sums   = (float*)ws;  ws += (size_t)NGRAPH * EMB * 4;
  float* cnt    = (float*)ws;  ws += (size_t)NGRAPH * 4;
  // f16 images (16B-aligned)
  uintptr_t a = ((uintptr_t)ws + 255) & ~(uintptr_t)255;
  _Float16* h_node_h = (_Float16*)a;           a += (size_t)N * EMB * 2;
  _Float16* atom_wT  = (_Float16*)a;           a += (size_t)EMB * EMB * 2;
  _Float16* emb_wT   = (_Float16*)a;           a += (size_t)EMB * 32 * 2;
  _Float16* node_wT  = (_Float16*)a;           a += (size_t)NLAYER * EMB * EMB * 2;
  _Float16* w0T      = (_Float16*)a;           a += (size_t)NLAYER * EMB * 384 * 2;
  _Float16* whT      = (_Float16*)a;           a += (size_t)NLAYER * NHID * EMB * EMB * 2;
  a = (a + 255) & ~(uintptr_t)255;
  _Float16* h_edge_h = (_Float16*)a;           a += (size_t)E * EMB * 2;
  a = (a + 255) & ~(uintptr_t)255;
  float* h_edge = (float*)a;

  // ---- weight prep: fp32 -> transposed f16 images ----
  transpose_w_kernel<<<(EMB * EMB + 255) / 256, 256, 0, stream>>>(atom_w, atom_wT, EMB);
  transpose_w_kernel<<<(32 * EMB + 255) / 256, 256, 0, stream>>>(edge_emb_w, emb_wT, 32);
  for (int l = 0; l < NLAYER; ++l) {
    transpose_w_kernel<<<(EMB * EMB + 255) / 256, 256, 0, stream>>>(
        node_w + (long)l * EMB * EMB, node_wT + (long)l * EMB * EMB, EMB);
    transpose_w_kernel<<<(384 * EMB + 255) / 256, 256, 0, stream>>>(
        edge_w0 + (long)l * 384 * EMB, w0T + (long)l * EMB * 384, 384);
    for (int k = 0; k < NHID; ++k)
      transpose_w_kernel<<<(EMB * EMB + 255) / 256, 256, 0, stream>>>(
          edge_w + ((long)l * NHID + k) * EMB * EMB,
          whT + ((long)l * NHID + k) * EMB * EMB, EMB);
  }

  int nodeBlocks = (N + 63) / 64;
  int embBlocks  = (E + 63) / 64;
  int mlpBlocks  = (E + EPB - 1) / EPB;

  node_gemm_kernel<<<nodeBlocks, 256, 0, stream>>>(atom_table, z, atom_wT, atom_b,
                                                   h_node, h_node_h, N, 0);
  edge_embed_kernel<<<embBlocks, 256, 0, stream>>>(pos, row, col, emb_wT, edge_emb_b,
                                                   h_edge, h_edge_h, E);

  long aggN = (long)N * EMB;
  for (int l = 0; l < NLAYER; ++l) {
    zero_kernel<<<(int)((aggN + 255) / 256), 256, 0, stream>>>(agg, aggN);
    edge_mlp_kernel<<<mlpBlocks, 256, 0, stream>>>(
        h_node_h, h_edge, h_edge_h, row, col,
        w0T + (long)l * EMB * 384, edge_b0 + l * EMB,
        whT + (long)l * NHID * EMB * EMB, edge_b + l * NHID * EMB,
        agg, E);
    node_gemm_kernel<<<nodeBlocks, 256, 0, stream>>>(
        agg, nullptr, node_wT + (long)l * EMB * EMB, node_b + l * EMB,
        h_node, h_node_h, N, 1);
  }

  long poolN = (long)NGRAPH * EMB + NGRAPH;   // sums + cnt contiguous
  zero_kernel<<<(int)((poolN + 255) / 256), 256, 0, stream>>>(sums, poolN);
  pool_kernel<<<(int)(((long)N * 4 + 255) / 256), 256, 0, stream>>>(h_node, batch,
                                                                    sums, cnt, N);
  out_kernel<<<1, 256, 0, stream>>>(sums, cnt, out_w, out_b, (float*)d_out);
}
